// CausalSelfAttention_87978110091517
// MI455X (gfx1250) — matrix operand order repaired and checked by reading, not compiled, not run on hardware
//
#include <hip/hip_runtime.h>
#include <hip/hip_bf16.h>

// ---------------------------------------------------------------------------
// CausalSelfAttention for MI455X (gfx1250, wave32, WMMA)
//   B=2, T=2048, C=1024, H=16, HD=64, BLOCK_SIZE=2048
// - all GEMM-class math via v_wmma_f32_16x16x32_f16 (f32 accumulate)
// - f16 intermediates (qkv, y) -> half the HBM traffic of f32
// - attention Q/K tiles staged with global_load_async_to_lds_b128 (ASYNCcnt)
// - LDS layouts give contiguous 16B fragment loads (ds_load_b128)
// - softmax parallelized 4 threads/row with wave32 shfl_xor reductions
// ---------------------------------------------------------------------------

typedef __attribute__((ext_vector_type(16))) _Float16 v16h;
typedef __attribute__((ext_vector_type(8)))  _Float16 v8h;
typedef __attribute__((ext_vector_type(4)))  _Float16 v4h;
typedef __attribute__((ext_vector_type(8)))  float    v8f;

constexpr int Bc  = 2;
constexpr int Tc  = 2048;
constexpr int Cc  = 1024;
constexpr int HDc = 64;
constexpr int BSz = 2048;          // BLOCK_SIZE
constexpr int MT  = Bc * Tc;       // 4096 rows

__device__ inline v8f vzero8() {
  v8f z = {0.f, 0.f, 0.f, 0.f, 0.f, 0.f, 0.f, 0.f};
  return z;
}

__device__ inline v8f wmma_f16(v16h a, v16h b, v8f c) {
  return __builtin_amdgcn_wmma_f32_16x16x32_f16(
      false, a, false, b, (short)0, c, false, false);
}

// A fragment (16x32 f16, ISA layout): two contiguous 16B LDS loads.
__device__ inline v16h frag_a(const _Float16* row, int kbase, int half) {
  const v8h lo = *(const v8h*)(row + kbase + half * 8);
  const v8h hi = *(const v8h*)(row + kbase + 16 + half * 8);
  return __builtin_shufflevector(lo, hi, 0, 1, 2, 3, 4, 5, 6, 7,
                                 8, 9, 10, 11, 12, 13, 14, 15);
}

// B fragment (32x16 f16, ISA layout): two contiguous 16B LDS loads (K-major column storage).
__device__ inline v16h frag_b(const _Float16* col, int kbase, int half) {
  const v8h lo = *(const v8h*)(col + kbase + half * 8);
  const v8h hi = *(const v8h*)(col + kbase + 16 + half * 8);
  return __builtin_shufflevector(lo, hi, 0, 1, 2, 3, 4, 5, 6, 7,
                                 8, 9, 10, 11, 12, 13, 14, 15);
}

// Async global -> LDS 16-byte copy (gfx1250 ASYNCcnt path).
__device__ inline void async_g2l_b128(unsigned lds_off, const void* g) {
  asm volatile("global_load_async_to_lds_b128 %0, %1, off"
               :: "v"(lds_off), "v"(g) : "memory");
}
__device__ inline void wait_async0() {
#if __has_builtin(__builtin_amdgcn_s_wait_asynccnt)
  __builtin_amdgcn_s_wait_asynccnt(0);
#else
  asm volatile("s_wait_asynccnt 0" ::: "memory");
#endif
}

// staging helpers: 4 elements -> f16 LDS
__device__ inline void stage4(_Float16* dst, const float* src) {
  float4 v = *(const float4*)src;
  dst[0] = (_Float16)v.x; dst[1] = (_Float16)v.y;
  dst[2] = (_Float16)v.z; dst[3] = (_Float16)v.w;
}
__device__ inline void stage4(_Float16* dst, const _Float16* src) {
  *(v4h*)dst = *(const v4h*)src;
}
__device__ inline void store_out(float* p, float v)     { *p = v; }
__device__ inline void store_out(_Float16* p, float v)  { *p = (_Float16)v; }

// ---------------------------------------------------------------------------
// Tiled WMMA GEMM:  Cout[M,N] = A[M,K] * W[K,N] + bias[N]
// A: f32 or f16 (row-major), W/bias: f32, Cout: f32 or f16.
// Block tile 64x128, K-step 32, double-buffered LDS; 8 waves = 4(M) x 2(N).
// ---------------------------------------------------------------------------
template <typename TA, typename TOut>
__global__ __launch_bounds__(256)
void gemm_wmma(const TA* __restrict__ A, const float* __restrict__ W,
               const float* __restrict__ bias, TOut* __restrict__ Cout,
               int M, int N, int K) {
  constexpr int BM = 64, BN = 128, BK = 32;
  constexpr int LDA = BK + 8;              // 80B rows, 16B-aligned
  __shared__ _Float16 As[2][BM][LDA];      // row-major (K contiguous)
  __shared__ _Float16 BsT[2][BN][LDA];     // N-major  (K contiguous)

  const int tid  = threadIdx.x;
  const int lane = tid & 31;
  const int half = lane >> 4, mr = lane & 15;
  const int wm = (tid >> 5) >> 1, wn = (tid >> 5) & 1;
  const int rowBase = blockIdx.y * BM;
  const int colBase = blockIdx.x * BN;

  v8f acc[4];
#pragma unroll
  for (int i = 0; i < 4; ++i) acc[i] = vzero8();

  auto load_tile = [&](int buf, int k0) {
#pragma unroll
    for (int it = 0; it < (BM * BK / 4) / 256; ++it) {     // 2 iters
      int i = tid + it * 256;
      int r = i >> 3, c4 = (i & 7) << 2;
      stage4(&As[buf][r][c4], &A[(size_t)(rowBase + r) * K + k0 + c4]);
    }
#pragma unroll
    for (int it = 0; it < (BK * BN / 4) / 256; ++it) {     // 4 iters
      int i = tid + it * 256;
      int r = i >> 5, c4 = (i & 31) << 2;
      float4 v = *(const float4*)&W[(size_t)(k0 + r) * N + colBase + c4];
      BsT[buf][c4 + 0][r] = (_Float16)v.x;
      BsT[buf][c4 + 1][r] = (_Float16)v.y;
      BsT[buf][c4 + 2][r] = (_Float16)v.z;
      BsT[buf][c4 + 3][r] = (_Float16)v.w;
    }
  };

  load_tile(0, 0);
  int cur = 0;
  for (int k0 = 0; k0 < K; k0 += BK) {
    __syncthreads();
    if (k0 + BK < K) load_tile(cur ^ 1, k0 + BK);
    if (k0 + 2 * BK < K) {
      __builtin_prefetch(&A[(size_t)(rowBase + (tid >> 2)) * K + k0 + 2 * BK +
                            ((tid & 3) << 3)], 0, 1);
      __builtin_prefetch(&W[(size_t)(k0 + 2 * BK + (tid >> 5)) * N + colBase +
                            ((tid & 31) << 2)], 0, 1);
    }
    // gather all fragments first, then issue WMMAs back-to-back
    v16h a = frag_a(&As[cur][wm * 16 + mr][0], 0, half);
    v16h bf[4];
#pragma unroll
    for (int nt = 0; nt < 4; ++nt)
      bf[nt] = frag_b(&BsT[cur][wn * 64 + nt * 16 + mr][0], 0, half);
#pragma unroll
    for (int nt = 0; nt < 4; ++nt) acc[nt] = wmma_f16(a, bf[nt], acc[nt]);
    cur ^= 1;
  }

#pragma unroll
  for (int nt = 0; nt < 4; ++nt) {
#pragma unroll
    for (int v = 0; v < 8; ++v) {
      int r = rowBase + wm * 16 + v + half * 8;
      int c = colBase + wn * 64 + nt * 16 + mr;
      store_out(&Cout[(size_t)r * N + c], acc[nt][v] + bias[c]);
    }
  }
}

// ---------------------------------------------------------------------------
// Histogram (cross-batch, matching the reference's scatter semantics) +
// per-batch inclusive scan of 1/(count+1e-10) gathered by token_index.
// ---------------------------------------------------------------------------
__global__ __launch_bounds__(1024)
void prep_scan(const int* __restrict__ token_index, float* __restrict__ prot) {
  __shared__ int   hist[BSz];
  __shared__ float buf0[Tc];
  __shared__ float buf1[Tc];
  const int tid = threadIdx.x;

  for (int i = tid; i < BSz; i += 1024) hist[i] = 0;
  __syncthreads();
  for (int i = tid; i < Bc * Tc; i += 1024)
    atomicAdd(&hist[token_index[i] & (BSz - 1)], 1);
  __syncthreads();

  for (int b = 0; b < Bc; ++b) {
    for (int i = tid; i < Tc; i += 1024) {
      int idx = token_index[b * Tc + i] & (BSz - 1);
      buf0[i] = 1.0f / ((float)hist[idx] + 1e-10f);
    }
    __syncthreads();
    float* src = buf0;
    float* dst = buf1;
    for (int off = 1; off < Tc; off <<= 1) {
      for (int i = tid; i < Tc; i += 1024)
        dst[i] = (i >= off) ? (src[i] + src[i - off]) : src[i];
      __syncthreads();
      float* t = src; src = dst; dst = t;
    }
    for (int i = tid; i < Tc; i += 1024) prot[b * Tc + i] = src[i];
    __syncthreads();
  }
}

// ---------------------------------------------------------------------------
// RoPE + fixups, in-place on f16 qkv [MT][3C]. One block per token; each
// thread owns a rotation pair (d, d+32) so in-place updates are race-free.
// ---------------------------------------------------------------------------
__global__ __launch_bounds__(256)
void rope_fix(_Float16* __restrict__ qkv, const float* __restrict__ prot,
              const float* __restrict__ cumsc,
              const unsigned char* __restrict__ padmask,
              const int* __restrict__ token_index,
              float* __restrict__ pm_tok) {
  const int bt = blockIdx.x;                // 0..MT-1
  const int b  = bt >> 11;                  // /Tc
  const float pr = prot[bt];
  const float cs = cumsc[bt];
  const float ev = __expf(cs);
  _Float16* row = qkv + (size_t)bt * (3 * Cc);

  for (int p = threadIdx.x; p < Cc / 2; p += 256) {
    const int head = p >> 5;                // 32 pairs per head
    const int j    = p & 31;
    const int d1   = head * HDc + j;
    const int d2   = d1 + 32;
    const float inv_freq = __expf(-(2.0f * (float)j / 64.0f) * 9.2103403719761836f);
    const float ang = pr * inv_freq;
    const float c = __cosf(ang), s = __sinf(ang);

    float q1 = (float)row[d1], q2 = (float)row[d2];
    row[d1] = (_Float16)(q1 * c - q2 * s);
    row[d2] = (_Float16)(q2 * c + q1 * s);

    float k1 = (float)row[Cc + d1], k2 = (float)row[Cc + d2];
    row[Cc + d1] = (_Float16)(k1 * c - k2 * s);
    row[Cc + d2] = (_Float16)(k2 * c + k1 * s);

    row[2 * Cc + d1] = (_Float16)((float)row[2 * Cc + d1] * ev);
    row[2 * Cc + d2] = (_Float16)((float)row[2 * Cc + d2] * ev);

    if (j == 31) {                          // d2 == head*64 + 63
      row[d2]      = (_Float16)1.0f;        // q last dim -> 1
      row[Cc + d2] = (_Float16)cs;          // k last dim -> cumulative score
    }
  }
  if (threadIdx.x == 0) {
    int idx = token_index[bt] & (BSz - 1);
    pm_tok[bt] = padmask[b * BSz + idx] ? 1.0f : 0.0f;
  }
}

// ---------------------------------------------------------------------------
// Flash attention per (b, h, 64-query tile). 8 waves: 4(q) x 2(k/d).
// Q/K tiles: async global->LDS pure copies. V: transposed VGPR staging
// (index-based addressing keeps loads in the global address space).
// Softmax: 4 threads/row + shfl_xor reductions. Causal chunk skipping.
// ---------------------------------------------------------------------------
__global__ __launch_bounds__(256)
void attn_wmma(const _Float16* __restrict__ qkv,
               const float* __restrict__ pm_tok, _Float16* __restrict__ Y) {
  constexpr int LDH = 72;                   // 144B rows, 16B-aligned
  const int bh  = blockIdx.y;               // 0..31
  const int b   = bh >> 4, h = bh & 15;
  const int qb0 = blockIdx.x * 64;
  const int tid = threadIdx.x;
  const int lane = tid & 31;
  const int half = lane >> 4, mr = lane & 15;
  const int wm = (tid >> 5) >> 1, wn = (tid >> 5) & 1;

  __shared__ _Float16 Qs[64][LDH];          // [q][d]
  __shared__ _Float16 Ks[64][LDH];          // [k][d]  (B-frag for S: col=k)
  __shared__ _Float16 VsT[64][LDH];         // [d][k]  (B-frag for PV: col=d)
  __shared__ float    Sl[64][68];           // masked logits
  __shared__ _Float16 Pl[64][LDH];          // exp probs, A of PV
  __shared__ float row_m[64], row_l[64], row_alpha[64];
  __shared__ float pmq[64], pmk[64];

  const size_t rs   = 3 * Cc;
  const int    qoff = h * HDc;
  const int    koff = Cc + h * HDc;
  const int    voff = 2 * Cc + h * HDc;
  const float  NEGINF = -__builtin_inff();
  const float  SCALE  = 0.125f;             // 1/sqrt(64)
  const size_t cstep  = (size_t)64 * rs;    // halves per K/V chunk

  // ---- async-copy Q tile (64 x 64 f16 = 512 x 16B) ----
#pragma unroll
  for (int it = 0; it < 2; ++it) {
    int i = tid + it * 256;
    int r = i >> 3, c8 = (i & 7) << 3;      // 8 halves per 16B
    async_g2l_b128((unsigned)(uintptr_t)&Qs[r][c8],
                   qkv + (size_t)(b * Tc + qb0 + r) * rs + qoff + c8);
  }
  if (tid < 64) {
    row_m[tid] = NEGINF;
    row_l[tid] = 0.f;
    pmq[tid]   = pm_tok[b * Tc + qb0 + tid];
  }

  // ---- hoisted per-thread element indices (advance by cstep per chunk) ----
  size_t   kgi[2];
  unsigned klds[2];
#pragma unroll
  for (int it = 0; it < 2; ++it) {
    int i = tid + it * 256;
    int r = i >> 3, c8 = (i & 7) << 3;
    kgi[it]  = (size_t)(b * Tc + r) * rs + koff + c8;
    klds[it] = (unsigned)(uintptr_t)&Ks[r][c8];
  }
  size_t vgi[4];
  int vr[4], vd4[4];
#pragma unroll
  for (int it = 0; it < 4; ++it) {
    int i = tid + it * 256;
    vr[it]  = i >> 4;
    vd4[it] = (i & 15) << 2;
    vgi[it] = (size_t)(b * Tc + vr[it]) * rs + voff + vd4[it];
  }
  const float* pmkp = pm_tok + b * Tc;

  v8f o0 = vzero8(), o1 = vzero8();
  const int nchunks = qb0 / 64 + 1;

  for (int kc = 0; kc < nchunks; ++kc) {
    __syncthreads();                        // prior readers of Ks/VsT/Pl done
    // K tile: async pure copy (global -> LDS, ASYNCcnt)
#pragma unroll
    for (int it = 0; it < 2; ++it) async_g2l_b128(klds[it], qkv + kgi[it]);
    // V tile: batch the 4 global loads, then the transposed LDS stores
    v4h vv[4];
#pragma unroll
    for (int it = 0; it < 4; ++it) vv[it] = *(const v4h*)(qkv + vgi[it]);
#pragma unroll
    for (int it = 0; it < 4; ++it) {
      VsT[vd4[it] + 0][vr[it]] = vv[it][0];
      VsT[vd4[it] + 1][vr[it]] = vv[it][1];
      VsT[vd4[it] + 2][vr[it]] = vv[it][2];
      VsT[vd4[it] + 3][vr[it]] = vv[it][3];
    }
    if (tid < 64) pmk[tid] = pmkp[kc * 64 + tid];
    if (kc + 1 < nchunks) {                 // prefetch next chunk
      __builtin_prefetch(qkv + kgi[0] + cstep, 0, 1);
      __builtin_prefetch(qkv + kgi[1] + cstep, 0, 1);
      __builtin_prefetch(qkv + vgi[0] + cstep, 0, 1);
      __builtin_prefetch(qkv + vgi[2] + cstep, 0, 1);
    }
#pragma unroll
    for (int it = 0; it < 2; ++it) kgi[it] += cstep;
#pragma unroll
    for (int it = 0; it < 4; ++it) vgi[it] += cstep;
    wait_async0();                          // own async stores landed
    __syncthreads();                        // everyone's stores visible

    // ---- S = Q @ K^T ----
    v8f s0 = vzero8(), s1 = vzero8();
#pragma unroll
    for (int kd = 0; kd < HDc; kd += 32) {
      v16h a  = frag_a(&Qs[wm * 16 + mr][0], kd, half);
      v16h b0 = frag_b(&Ks[wn * 32 + mr][0], kd, half);
      v16h b1 = frag_b(&Ks[wn * 32 + 16 + mr][0], kd, half);
      s0 = wmma_f16(a, b0, s0);
      s1 = wmma_f16(a, b1, s1);
    }
    // scaled + masked write to LDS
#pragma unroll
    for (int v = 0; v < 8; ++v) {
      int rl = wm * 16 + v + half * 8;
      int qg = qb0 + rl;
      int kl0 = wn * 32 + mr, kl1 = kl0 + 16;
      int kg0 = kc * 64 + kl0, kg1 = kc * 64 + kl1;
      float va = s0[v] * SCALE, vb = s1[v] * SCALE;
      if (kg0 > qg || pmq[rl] * pmk[kl0] == 0.f) va = NEGINF;
      if (kg1 > qg || pmq[rl] * pmk[kl1] == 0.f) vb = NEGINF;
      Sl[rl][kl0] = va;
      Sl[rl][kl1] = vb;
    }
    __syncthreads();

    // ---- online softmax: 4 threads per row, shfl_xor reductions ----
    {
      const int r = tid >> 2;               // 0..63
      const int c = (tid & 3) << 4;         // 0,16,32,48
      const float mold = row_m[r];
      float mc = mold;
#pragma unroll
      for (int j = 0; j < 16; ++j) mc = fmaxf(mc, Sl[r][c + j]);
      mc = fmaxf(mc, __shfl_xor(mc, 1));
      mc = fmaxf(mc, __shfl_xor(mc, 2));
      float alpha = (mc == mold) ? 1.0f : __expf(mold - mc);
      float sum = 0.f;
#pragma unroll
      for (int j = 0; j < 16; ++j) {
        float sv = Sl[r][c + j];
        float p = (sv == NEGINF) ? 0.f : __expf(sv - mc);
        sum += p;
        Pl[r][c + j] = (_Float16)p;
      }
      sum += __shfl_xor(sum, 1);
      sum += __shfl_xor(sum, 2);
      if ((tid & 3) == 0) {
        row_m[r] = mc;
        row_l[r] = row_l[r] * alpha + sum;
        row_alpha[r] = alpha;
      }
    }
    __syncthreads();

    // ---- O = O*alpha + P @ V ----
#pragma unroll
    for (int v = 0; v < 8; ++v) {
      float al = row_alpha[wm * 16 + v + half * 8];
      o0[v] *= al;
      o1[v] *= al;
    }
#pragma unroll
    for (int kd = 0; kd < 64; kd += 32) {
      v16h a  = frag_a(&Pl[wm * 16 + mr][0], kd, half);
      v16h b0 = frag_b(&VsT[wn * 32 + mr][0], kd, half);
      v16h b1 = frag_b(&VsT[wn * 32 + 16 + mr][0], kd, half);
      o0 = wmma_f16(a, b0, o0);
      o1 = wmma_f16(a, b1, o1);
    }
  }
  __syncthreads();

  // ---- normalize and write y[b, t, h*64 + d] (f16) ----
#pragma unroll
  for (int v = 0; v < 8; ++v) {
    int rl = wm * 16 + v + half * 8;
    float inv_l = 1.0f / row_l[rl];
    int c0 = wn * 32 + mr, c1 = c0 + 16;
    size_t yrow = (size_t)(b * Tc + qb0 + rl) * Cc + h * HDc;
    Y[yrow + c0] = (_Float16)(o0[v] * inv_l);
    Y[yrow + c1] = (_Float16)(o1[v] * inv_l);
  }
}

// ---------------------------------------------------------------------------
extern "C" void kernel_launch(void* const* d_in, const int* in_sizes, int n_in,
                              void* d_out, int out_size, void* d_ws, size_t ws_size,
                              hipStream_t stream) {
  const float*         x     = (const float*)d_in[0];
  const float*         cum   = (const float*)d_in[1];
  const unsigned char* pmask = (const unsigned char*)d_in[2];   // bool_
  const int*           tok   = (const int*)d_in[3];
  const float*         Wattn = (const float*)d_in[4];
  const float*         battn = (const float*)d_in[5];
  const float*         Wproj = (const float*)d_in[6];
  const float*         bproj = (const float*)d_in[7];
  float*               out   = (float*)d_out;

  float*     prot  = (float*)d_ws;                       // MT f32
  float*     pmtok = prot + MT;                          // MT f32
  _Float16*  qkv   = (_Float16*)(pmtok + MT);            // MT * 3C f16
  _Float16*  ybuf  = qkv + (size_t)MT * 3 * Cc;          // MT * C  f16

  // 1) qkv = x @ W_attn + b_attn        [4096 x 3072] (f32 in, f16 out)
  gemm_wmma<float, _Float16><<<dim3(3 * Cc / 128, MT / 64), 256, 0, stream>>>(
      x, Wattn, battn, qkv, MT, 3 * Cc, Cc);

  // 2) token histogram + cumulative partial rotations
  prep_scan<<<1, 1024, 0, stream>>>(tok, prot);

  // 3) RoPE + last-dim fixups + v scaling + padding-mask gather (f16 in-place)
  rope_fix<<<MT, 256, 0, stream>>>(qkv, prot, cum, pmask, tok, pmtok);

  // 4) flash attention (WMMA QK^T and PV, async Q/K staging)
  attn_wmma<<<dim3(Tc / 64, Bc * 16), 256, 0, stream>>>(qkv, pmtok, ybuf);

  // 5) out = y @ W_proj + b_proj        [4096 x 1024] (f16 in, f32 out)
  gemm_wmma<_Float16, float><<<dim3(Cc / 128, MT / 64), 256, 0, stream>>>(
      ybuf, Wproj, bproj, out, MT, Cc, Cc);
}
